// PointPillars_16793322127974
// MI455X (gfx1250) — matrix operand order repaired and checked by reading, not compiled, not run on hardware
//
#include <hip/hip_runtime.h>
#include <hip/hip_bf16.h>

typedef __bf16 bf16_t;
typedef __attribute__((ext_vector_type(16))) __bf16 v16bf;
typedef __attribute__((ext_vector_type(8)))  __bf16 v8bf;
typedef __attribute__((ext_vector_type(8)))  float  v8f;

#define EPSV 1e-5f

// ---- problem dims ----
constexpr int B_    = 4;
constexpr int P_    = 12000;
constexpr int NPT   = 32;      // points per pillar
constexpr int CPF   = 9;       // raw point features
constexpr int NY_   = 496;
constexpr int NX_   = 432;
constexpr int H2_   = 248;
constexpr int W2_   = 216;
constexpr int HW2_  = H2_ * W2_;        // 53568
constexpr int MTOT_ = B_ * HW2_;        // 214272 output pixels
constexpr int MTILES_ = MTOT_ / 16;     // 13392 (exact)

// ---- workspace layout (bytes, 256-aligned) ----
constexpr size_t OFF_PF  = 0;                               // [B,P,64] bf16   = 6,144,000
constexpr size_t OFF_CAN = OFF_PF  + 6144000ull;            // [B,496,432,64] bf16 = 109,707,264
constexpr size_t OFF_X1  = OFF_CAN + 109707264ull;          // [B,248,216,64] bf16 = 27,426,816
constexpr size_t OFF_X2  = OFF_X1  + 27426816ull;           // [B,248,216,128] bf16 = 54,853,632
constexpr size_t OFF_W1P = OFF_X2  + 54853632ull;           // [64][576] bf16
constexpr size_t OFF_W2P = OFF_W1P + 73728ull;              // [128][576] bf16
constexpr size_t OFF_WHP = OFF_W2P + 147456ull;             // [32][128] bf16 (heads, padded)
constexpr size_t OFF_WHB = OFF_WHP + 8192ull;               // [32] f32 head biases

// ------------------------------------------------------------------
// zero fill (16B granularity)
__global__ void zero16_kernel(uint4* __restrict__ p, long n) {
  long i = (long)blockIdx.x * blockDim.x + threadIdx.x;
  if (i < n) p[i] = uint4{0u, 0u, 0u, 0u};
}

// ------------------------------------------------------------------
// Repack OIHW fp32 conv weights -> [Cout][k] bf16, k = (ky*3+kx)*Cin + ci
__global__ void pack_conv_w(const float* __restrict__ w, bf16_t* __restrict__ wp,
                            int Cout, int Cin) {
  const int Ktot = Cin * 9;
  const int i = blockIdx.x * blockDim.x + threadIdx.x;
  if (i >= Cout * Ktot) return;
  const int co = i / Ktot, k = i % Ktot;
  const int tap = k / Cin, ci = k % Cin;
  const int ky = tap / 3, kx = tap % 3;
  wp[i] = (bf16_t)w[(((size_t)co * Cin + ci) * 3 + ky) * 3 + kx];
}

// Pack head weights [cls(6)|reg(14)|dir(4)|pad(8)] -> [32][128] bf16 + bias[32]
__global__ void pack_heads(const float* __restrict__ cw, const float* __restrict__ cb,
                           const float* __restrict__ rw, const float* __restrict__ rb,
                           const float* __restrict__ dw, const float* __restrict__ db,
                           bf16_t* __restrict__ whp, float* __restrict__ whb) {
  const int i = blockIdx.x * blockDim.x + threadIdx.x;
  if (i < 32) {
    float bv = 0.f;
    if (i < 6)       bv = cb[i];
    else if (i < 20) bv = rb[i - 6];
    else if (i < 24) bv = db[i - 20];
    whb[i] = bv;
  }
  if (i >= 32 * 128) return;
  const int co = i / 128, c = i % 128;
  float v = 0.f;
  if (co < 6)       v = cw[co * 128 + c];
  else if (co < 20) v = rw[(co - 6) * 128 + c];
  else if (co < 24) v = dw[(co - 20) * 128 + c];
  whp[i] = (bf16_t)v;
}

// ------------------------------------------------------------------
// PillarFeatureNet: per pillar, feat[o] = max_n relu(bn(dot9(pillar[n], w[o])))
// one block (64 threads) per (b,p); pillar staged in LDS so raw points are read once.
__global__ void pfn_kernel(const float* __restrict__ pillars,
                           const float* __restrict__ w,
                           const float* __restrict__ g,  const float* __restrict__ bb,
                           const float* __restrict__ mm, const float* __restrict__ vv,
                           bf16_t* __restrict__ pf) {
  __shared__ float s[NPT * CPF];
  const int bp = blockIdx.x;
  const float* src = pillars + (size_t)bp * NPT * CPF;
  for (int i = threadIdx.x; i < NPT * CPF; i += 64) s[i] = src[i];
  __syncthreads();
  const int o = threadIdx.x;
  float wr[CPF];
#pragma unroll
  for (int c = 0; c < CPF; ++c) wr[c] = w[o * CPF + c];
  const float sc = g[o] * rsqrtf(vv[o] + EPSV);
  const float sh = bb[o] - mm[o] * sc;
  float best = 0.f;  // relu output >= 0 always
  for (int n = 0; n < NPT; ++n) {
    float d = 0.f;
#pragma unroll
    for (int c = 0; c < CPF; ++c) d += s[n * CPF + c] * wr[c];
    best = fmaxf(best, fmaxf(d * sc + sh, 0.f));
  }
  pf[(size_t)bp * 64 + o] = (bf16_t)best;
}

// Scatter pillar features into zeroed NHWC canvas (unique coords, drop (0,0))
__global__ void scatter_kernel(const bf16_t* __restrict__ pf,
                               const int* __restrict__ coords,
                               bf16_t* __restrict__ canvas) {
  const int bp = blockIdx.x;
  const int x = coords[bp * 2 + 0];
  const int y = coords[bp * 2 + 1];
  if (x + y == 0) return;  // reference: coords.sum(-1) != 0
  const int b = bp / P_;
  canvas[(((size_t)b * NY_ + y) * NX_ + x) * 64 + threadIdx.x] =
      pf[(size_t)bp * 64 + threadIdx.x];
}

// ------------------------------------------------------------------
// Implicit-GEMM 3x3 conv (pad 1), bf16 WMMA, NHWC activations, weights [Cout][9*Cin].
// Each wave: 32 output pixels x 64 output channels (2 M-tiles x 4 N-tiles);
// each B fragment is reused by two WMMAs. Fully unrolled K loop (compile-time dims)
// keeps accumulators pinned in fixed VGPRs. BN+ReLU fused on f32 accumulators.
template <int CIN, int COUT, int STRIDE, int HIN, int WIN, int HOUT, int WOUT>
__global__ __launch_bounds__(256) void conv3x3_bf16_wmma(
    const bf16_t* __restrict__ in, const bf16_t* __restrict__ wp,
    bf16_t* __restrict__ out,
    const float* __restrict__ bng, const float* __restrict__ bnb,
    const float* __restrict__ bnm, const float* __restrict__ bnv) {
  constexpr int KTOT  = CIN * 9;
  constexpr int HWOUT = HOUT * WOUT;
  constexpr int CPT   = CIN / 32;                 // K-chunks per tap
  constexpr int MT2   = (B_ * HWOUT) / 32;        // 32-row M tiles (exact)
  constexpr int NWAVES = MT2 * (COUT / 64);

  const int wave = (blockIdx.x * blockDim.x + threadIdx.x) >> 5;
  if (wave >= NWAVES) return;  // wave-uniform: EXEC all-ones for WMMA
  const int lane = threadIdx.x & 31;
  const int row  = lane & 15;   // A row / B,C,D column index
  const int half = lane >> 4;   // K sub-range selector
  const int mtile = wave % MT2;
  const int n0    = (wave / MT2) * 64;
  const int m0    = mtile * 32;

  // decode the two A-row pixels owned by this lane
  const int mA = m0 + row;        // M-tile 0
  const int mB = m0 + 16 + row;   // M-tile 1
  const int bA = mA / HWOUT, remA = mA % HWOUT;
  const int bB = mB / HWOUT, remB = mB % HWOUT;
  const int oyA = remA / WOUT, oxA = remA % WOUT;
  const int oyB = remB / WOUT, oxB = remB % WOUT;

  v8f acc0[4] = {};
  v8f acc1[4] = {};

#pragma unroll
  for (int ky = 0; ky < 3; ++ky) {
    const int iyA = oyA * STRIDE + ky - 1;
    const int iyB = oyB * STRIDE + ky - 1;
    const bool yokA = (iyA >= 0) && (iyA < HIN);
    const bool yokB = (iyB >= 0) && (iyB < HIN);
#pragma unroll
    for (int kx = 0; kx < 3; ++kx) {
      const int ixA = oxA * STRIDE + kx - 1;
      const int ixB = oxB * STRIDE + kx - 1;
      const bool okA = yokA && (ixA >= 0) && (ixA < WIN);
      const bool okB = yokB && (ixB >= 0) && (ixB < WIN);
      const bf16_t* ibaseA = in + (((size_t)bA * HIN + iyA) * WIN + ixA) * CIN;
      const bf16_t* ibaseB = in + (((size_t)bB * HIN + iyB) * WIN + ixB) * CIN;
      const int tap = ky * 3 + kx;
#pragma unroll
      for (int hb = 0; hb < CPT; ++hb) {
        const int cb = hb * 32;
        // A layout (16-bit, 16x32): half 0 -> k {0..15}, half 1 -> k {16..31}
        v8bf a0r1 = {}, a0r2 = {}, a1r1 = {}, a1r2 = {};
        if (okA) {
          a0r1 = *(const v8bf*)(ibaseA + cb + 16 * half);
          a0r2 = *(const v8bf*)(ibaseA + cb + 16 * half + 8);
        }
        if (okB) {
          a1r1 = *(const v8bf*)(ibaseB + cb + 16 * half);
          a1r2 = *(const v8bf*)(ibaseB + cb + 16 * half + 8);
        }
        v16bf a0 = __builtin_shufflevector(a0r1, a0r2, 0, 1, 2, 3, 4, 5, 6, 7,
                                           8, 9, 10, 11, 12, 13, 14, 15);
        v16bf a1 = __builtin_shufflevector(a1r1, a1r2, 0, 1, 2, 3, 4, 5, 6, 7,
                                           8, 9, 10, 11, 12, 13, 14, 15);
        const int kchunk = tap * CPT + hb;
        const bf16_t* wbase = wp + (size_t)kchunk * 32 + 16 * half;
#pragma unroll
        for (int t = 0; t < 4; ++t) {
          const int co = n0 + t * 16 + row;
          // B layout (32x16): lane element i -> k = i + 16*half, col = lane%16
          v16bf bv = *(const v16bf*)(wbase + (size_t)co * KTOT);
          acc0[t] = __builtin_amdgcn_wmma_f32_16x16x32_bf16(
              false, a0, false, bv, (short)0, acc0[t], false, false);
          acc1[t] = __builtin_amdgcn_wmma_f32_16x16x32_bf16(
              false, a1, false, bv, (short)0, acc1[t], false, false);
        }
      }
    }
  }

  // BN + ReLU, store bf16 NHWC
  float sct[4], sht[4];
#pragma unroll
  for (int t = 0; t < 4; ++t) {
    const int co = n0 + t * 16 + row;
    sct[t] = bng[co] * rsqrtf(bnv[co] + EPSV);
    sht[t] = bnb[co] - bnm[co] * sct[t];
  }
#pragma unroll
  for (int j = 0; j < 2; ++j) {
#pragma unroll
    for (int t = 0; t < 4; ++t) {
      const int co = n0 + t * 16 + row;
      const v8f* accp = j ? acc1 : acc0;
#pragma unroll
      for (int r = 0; r < 8; ++r) {
        const int mr   = m0 + j * 16 + r + 8 * half;  // C/D: M = r + 8*half
        const int br   = mr / HWOUT;
        const int remr = mr % HWOUT;
        float v = fmaxf(accp[t][r] * sct[t] + sht[t], 0.f);
        out[((size_t)br * HWOUT + remr) * COUT + co] = (bf16_t)v;
      }
    }
  }
}

// ------------------------------------------------------------------
// Detection heads: GEMM [M,128] x [128,32] (24 real channels + pad),
// bias add, scatter into concatenated NCHW fp32 output.
__global__ void heads_wmma(const bf16_t* __restrict__ x2,
                           const bf16_t* __restrict__ whp,
                           const float* __restrict__ whb,
                           float* __restrict__ out,
                           int total_waves) {
  const int wave = (blockIdx.x * blockDim.x + threadIdx.x) >> 5;
  if (wave >= total_waves) return;
  const int lane = threadIdx.x & 31;
  const int row  = lane & 15;
  const int half = lane >> 4;
  const int m = wave * 16 + row;
  const bf16_t* ibase = x2 + (size_t)m * 128;

  v8f acc[2] = {};
#pragma unroll
  for (int kchunk = 0; kchunk < 4; ++kchunk) {
    const int cb = kchunk * 32;
    v8bf r1 = *(const v8bf*)(ibase + cb + 16 * half);
    v8bf r2 = *(const v8bf*)(ibase + cb + 16 * half + 8);
    v16bf a = __builtin_shufflevector(r1, r2, 0, 1, 2, 3, 4, 5, 6, 7,
                                      8, 9, 10, 11, 12, 13, 14, 15);
#pragma unroll
    for (int t = 0; t < 2; ++t) {
      const int co = t * 16 + row;
      v16bf bv = *(const v16bf*)(whp + (size_t)co * 128 + cb + 16 * half);
      acc[t] = __builtin_amdgcn_wmma_f32_16x16x32_bf16(
          false, a, false, bv, (short)0, acc[t], false, false);
    }
  }

#pragma unroll
  for (int t = 0; t < 2; ++t) {
    const int co = t * 16 + row;
    if (co >= 24) continue;
    const float bias = whb[co];
#pragma unroll
    for (int r = 0; r < 8; ++r) {
      const int mr  = wave * 16 + r + 8 * half;
      const int b   = mr / HW2_;
      const int rem = mr % HW2_;
      const float v = acc[t][r] + bias;
      size_t idx;
      if (co < 6)
        idx = ((size_t)(b * 6 + co)) * HW2_ + rem;                        // cls [4,6,H,W]
      else if (co < 20)
        idx = (size_t)4 * 6 * HW2_ + ((size_t)(b * 14 + (co - 6))) * HW2_ + rem;   // reg
      else
        idx = (size_t)4 * 20 * HW2_ + ((size_t)(b * 4 + (co - 20))) * HW2_ + rem;  // dir
      out[idx] = v;
    }
  }
}

// ------------------------------------------------------------------
extern "C" void kernel_launch(void* const* d_in, const int* in_sizes, int n_in,
                              void* d_out, int out_size, void* d_ws, size_t ws_size,
                              hipStream_t stream) {
  (void)in_sizes; (void)n_in; (void)out_size; (void)ws_size;

  const float* pillars = (const float*)d_in[0];
  const int*   coords  = (const int*)d_in[1];
  // d_in[2] (num_points) is unused by the reference math
  const float* pfn_w = (const float*)d_in[3];
  const float* pfn_g = (const float*)d_in[4];
  const float* pfn_b = (const float*)d_in[5];
  const float* pfn_m = (const float*)d_in[6];
  const float* pfn_v = (const float*)d_in[7];
  const float* w1 = (const float*)d_in[8];
  const float* g1 = (const float*)d_in[9];
  const float* b1 = (const float*)d_in[10];
  const float* m1 = (const float*)d_in[11];
  const float* v1 = (const float*)d_in[12];
  const float* w2 = (const float*)d_in[13];
  const float* g2 = (const float*)d_in[14];
  const float* b2 = (const float*)d_in[15];
  const float* m2 = (const float*)d_in[16];
  const float* v2 = (const float*)d_in[17];
  const float* cls_w = (const float*)d_in[18];
  const float* cls_b = (const float*)d_in[19];
  const float* reg_w = (const float*)d_in[20];
  const float* reg_b = (const float*)d_in[21];
  const float* dir_w = (const float*)d_in[22];
  const float* dir_b = (const float*)d_in[23];

  char* ws = (char*)d_ws;
  bf16_t* pf     = (bf16_t*)(ws + OFF_PF);
  bf16_t* canvas = (bf16_t*)(ws + OFF_CAN);
  bf16_t* x1     = (bf16_t*)(ws + OFF_X1);
  bf16_t* x2     = (bf16_t*)(ws + OFF_X2);
  bf16_t* w1p    = (bf16_t*)(ws + OFF_W1P);
  bf16_t* w2p    = (bf16_t*)(ws + OFF_W2P);
  bf16_t* whp    = (bf16_t*)(ws + OFF_WHP);
  float*  whb    = (float*)(ws + OFF_WHB);

  // canvas = 109,707,264 B = 6,856,704 x 16B; 6,856,704 / 256 = 26784 blocks exact
  zero16_kernel<<<26784, 256, 0, stream>>>((uint4*)canvas, 6856704L);

  pack_conv_w<<<144, 256, 0, stream>>>(w1, w1p, 64, 64);    // 36864 elems
  pack_conv_w<<<288, 256, 0, stream>>>(w2, w2p, 128, 64);   // 73728 elems
  pack_heads<<<16, 256, 0, stream>>>(cls_w, cls_b, reg_w, reg_b, dir_w, dir_b, whp, whb);

  pfn_kernel<<<B_ * P_, 64, 0, stream>>>(pillars, pfn_w, pfn_g, pfn_b, pfn_m, pfn_v, pf);
  scatter_kernel<<<B_ * P_, 64, 0, stream>>>(pf, coords, canvas);

  // conv1: 64->64, stride 2: 6696 32-row M-tiles x 1 column group = 6696 waves (8/block)
  conv3x3_bf16_wmma<64, 64, 2, NY_, NX_, H2_, W2_>
      <<<(MTOT_ / 32) / 8, 256, 0, stream>>>(canvas, w1p, x1, g1, b1, m1, v1);

  // conv2: 64->128, stride 1: 6696 x 2 column groups = 13392 waves
  conv3x3_bf16_wmma<64, 128, 1, H2_, W2_, H2_, W2_>
      <<<(2 * (MTOT_ / 32)) / 8, 256, 0, stream>>>(x1, w2p, x2, g2, b2, m2, v2);

  // heads: 13392 waves
  heads_wmma<<<MTILES_ / 8, 256, 0, stream>>>(x2, whp, whb, (float*)d_out, MTILES_);
}
